// LearnableMaskTemporalAggregation_7138235646316
// MI455X (gfx1250) — matrix operand order repaired and checked by reading, not compiled, not run on hardware
//
#include <hip/hip_runtime.h>
#include <hip/hip_bf16.h>

// ---------------------------------------------------------------------------
// out[b,n,j] = sum_m softmax_row(pred[n,:] + off[j,n,:])[m] * x[b,m,j]
// x [64,512,96] f32, off [96,512,512] f32, pred [512,512] f32 -> out [64,512,96] f32
//
// Kernel 1: pre-pack x into bf16 K-pair layout xw[j][p][b] (p = m/2) in d_ws.
// Kernel 2: fused per-(j, 64-row n-tile): stream offsets once -> LDS softmax
//           -> v_wmma_f32_16x16x32_bf16 GEMM against LDS-staged x slice.
// LDS rows padded to kill bank conflicts (64 banks x 4B):
//   rowBuf stride 528 f32 (<=2-way), maskPk stride 260 u32 (2-way floor).
// ---------------------------------------------------------------------------

#define SEQ    512
#define JDIM   96
#define BATCH  64
#define ROWS   64          // n-rows per block
#define NTHR   512         // 16 waves
#define PAIRS  (SEQ / 2)   // 256 bf16 K-pairs per row
#define RB_STR 528         // rowBuf row stride in floats (512 + 16)
#define MK_STR 260         // maskPk row stride in u32    (256 + 4)

typedef __attribute__((ext_vector_type(16))) __bf16      v16bf;
typedef __attribute__((ext_vector_type(8)))  float       v8f;
typedef __attribute__((ext_vector_type(4)))  unsigned    v4u;
typedef __attribute__((ext_vector_type(8)))  unsigned    v8u;

#define RB_BYTES (ROWS * RB_STR * 4)          // 135168
#define MK_BYTES (ROWS * MK_STR * 4)          // 66560
#define XP_BYTES (PAIRS * BATCH * 4)          // 65536
#define SMEM_BYTES (RB_BYTES + MK_BYTES + XP_BYTES)   // 267264 (< 320 KB/WGP)

__device__ __forceinline__ unsigned pk_bf16(float a, float b) {
    __bf16 x = (__bf16)a;
    __bf16 y = (__bf16)b;
    unsigned short ux = __builtin_bit_cast(unsigned short, x);
    unsigned short uy = __builtin_bit_cast(unsigned short, y);
    return (unsigned)ux | ((unsigned)uy << 16);
}

// ---- Kernel 1: pack x[b, m, j] (f32) -> xw[j][p][b] (bf16 pair in u32) -----
__global__ __launch_bounds__(256)
void lmta_pack_x_kernel(const float* __restrict__ x, unsigned* __restrict__ xw) {
    // gid fastest over j -> coalesced reads of x (j is innermost in x)
    int gid = blockIdx.x * 256 + threadIdx.x;            // [0, 96*256*64)
    int j = gid % JDIM;
    int t = gid / JDIM;                                  // [0, 256*64)
    int p = t % PAIRS;
    int b = t / PAIRS;
    size_t base = ((size_t)b * SEQ + 2 * p) * JDIM + j;
    float f0 = x[base];
    float f1 = x[base + JDIM];
    xw[(size_t)j * (PAIRS * BATCH) + p * BATCH + b] = pk_bf16(f0, f1);
}

// ---- Kernel 2: fused softmax + WMMA GEMM -----------------------------------
__global__ __launch_bounds__(NTHR)
void lmta_fused_wmma_kernel(const unsigned* __restrict__ xw,
                            const float* __restrict__ off,
                            const float* __restrict__ pred,
                            float* __restrict__ out) {
    extern __shared__ __align__(16) char smem[];
    float*    rowBuf = (float*)smem;                          // 64 rows x 528 f32
    unsigned* maskPk = (unsigned*)(smem + RB_BYTES);          // 64 rows x 260 u32
    unsigned* xPk    = (unsigned*)(smem + RB_BYTES + MK_BYTES); // 256 x 64 u32

    const int tid = threadIdx.x;
    const int j   = blockIdx.y;            // joint index   [0,96)
    const int n0  = blockIdx.x * ROWS;     // n-tile origin [0,512) step 64

    // ---- Phase 1: stream pred + offset rows into LDS (100 MB tensor read
    //               exactly once, float4 + prefetch-ahead) -------------------
    {
        const float4* off4  = (const float4*)(off + (size_t)j * SEQ * SEQ + (size_t)n0 * SEQ);
        const float4* pred4 = (const float4*)(pred + (size_t)n0 * SEQ);
        float4*       rb4   = (float4*)rowBuf;
        #pragma unroll 4
        for (int idx = tid; idx < ROWS * SEQ / 4; idx += NTHR) {
            __builtin_prefetch((const char*)(off4 + idx) + 32768, 0, 1);
            float4 a = off4[idx];
            float4 c = pred4[idx];
            float4 r;
            r.x = a.x + c.x; r.y = a.y + c.y; r.z = a.z + c.z; r.w = a.w + c.w;
            int row = idx >> 7;            // 128 float4 per logical row
            int col = idx & 127;
            rb4[row * (RB_STR / 4) + col] = r;
        }
    }

    // ---- Phase 2: stage pre-packed x slice for this j (coalesced uint4) ----
    {
        const v4u* src = (const v4u*)(xw + (size_t)j * (PAIRS * BATCH));
        v4u*       dst = (v4u*)xPk;
        #pragma unroll 4
        for (int idx = tid; idx < PAIRS * BATCH / 4; idx += NTHR)
            dst[idx] = src[idx];
    }
    __syncthreads();

    // ---- Phase 3: row softmax (8 lanes per row, wave-synchronous) ----------
    {
        const int r = tid >> 3;          // row [0,64)
        const int s = tid & 7;           // sub-lane [0,8)
        float4* row4 = (float4*)(rowBuf + r * RB_STR);   // 128 live float4 per row

        float mx = -3.402823e38f;
        #pragma unroll
        for (int k = 0; k < 16; ++k) {
            float4 v = row4[s + k * 8];
            mx = fmaxf(mx, fmaxf(fmaxf(v.x, v.y), fmaxf(v.z, v.w)));
        }
        #pragma unroll
        for (int d = 1; d < 8; d <<= 1) mx = fmaxf(mx, __shfl_xor(mx, d, 32));

        float sum = 0.0f;
        #pragma unroll
        for (int k = 0; k < 16; ++k) {
            float4 v = row4[s + k * 8];
            v.x = __expf(v.x - mx); v.y = __expf(v.y - mx);
            v.z = __expf(v.z - mx); v.w = __expf(v.w - mx);
            sum += (v.x + v.y) + (v.z + v.w);
            row4[s + k * 8] = v;
        }
        #pragma unroll
        for (int d = 1; d < 8; d <<= 1) sum += __shfl_xor(sum, d, 32);
        float inv = __frcp_rn(sum);

        // normalize + pack bf16 pairs into WMMA-A staging buffer
        unsigned* mrow = maskPk + r * MK_STR;
        #pragma unroll
        for (int k = 0; k < 16; ++k) {
            int q = s + k * 8;           // float4 index -> pairs 2q, 2q+1
            float4 v = row4[q];
            mrow[2 * q]     = pk_bf16(v.x * inv, v.y * inv);
            mrow[2 * q + 1] = pk_bf16(v.z * inv, v.w * inv);
        }
    }
    __syncthreads();

    // ---- Phase 4: WMMA GEMM: 16 waves x one 16n x 16b tile, K = 512 --------
    {
        const int wave  = tid >> 5;      // [0,16)
        const int lane  = tid & 31;
        const int laneM = lane & 15;
        const int half  = lane >> 4;
        const int nt    = wave >> 2;     // n sub-tile [0,4)
        const int bt    = wave & 3;      // b sub-tile [0,4)
        const int b0    = bt * 16;

        const unsigned* aRow = maskPk + (nt * 16 + laneM) * MK_STR;
        const unsigned* bCol = xPk + b0 + laneM;

        v8f acc = {};
        #pragma unroll
        for (int step = 0; step < SEQ / 32; ++step) {
            const int kkp = step * 16;   // 16 bf16-pairs per K=32 chunk

            // A fragment: lanes 0-15 take K {0..7,16..23}, lanes 16-31 {8..15,24..31}
            v4u a0 = *(const v4u*)(aRow + kkp + half * 4);
            v4u a1 = *(const v4u*)(aRow + kkp + 8 + half * 4);
            v8u au; au.lo = a0; au.hi = a1;
            v16bf A = __builtin_bit_cast(v16bf, au);

            // B fragment: lane = N, lanes 0-15 K {0..7,16..23}, lanes 16-31 {8..15,24..31}
            v8u bu;
            #pragma unroll
            for (int v = 0; v < 8; ++v)
                bu[v] = bCol[(kkp + half * 4 + (v & 3) + (v >> 2) * 8) * BATCH];
            v16bf Bm = __builtin_bit_cast(v16bf, bu);

            acc = __builtin_amdgcn_wmma_f32_16x16x32_bf16(
                    /*neg_a=*/false, A, /*neg_b=*/false, Bm,
                    /*c_mod=*/(short)0, acc, /*reuse_a=*/false, /*reuse_b=*/false);
        }

        // D layout: VGPR g -> M = g + 8*half, N = lane&15
        const int b = b0 + laneM;
        #pragma unroll
        for (int g = 0; g < 8; ++g) {
            int n = n0 + nt * 16 + g + 8 * half;
            out[((size_t)b * SEQ + n) * JDIM + j] = acc[g];
        }
    }
}

extern "C" void kernel_launch(void* const* d_in, const int* in_sizes, int n_in,
                              void* d_out, int out_size, void* d_ws, size_t ws_size,
                              hipStream_t stream) {
    const float* x    = (const float*)d_in[0];   // [64, 512, 96]
    const float* off  = (const float*)d_in[1];   // [96, 512, 512]
    const float* pred = (const float*)d_in[2];   // [512, 512]
    float*       out  = (float*)d_out;           // [64, 512, 96]
    unsigned*    xw   = (unsigned*)d_ws;         // [96][256][64] u32 = 6 MB

    (void)in_sizes; (void)n_in; (void)out_size; (void)ws_size;

    // Kernel 1: pack x -> bf16 pairs, [j][p][b]
    {
        int total = JDIM * PAIRS * BATCH;        // 1,572,864
        lmta_pack_x_kernel<<<total / 256, 256, 0, stream>>>(x, xw);
    }

    // Kernel 2: fused softmax + WMMA GEMM
    {
        dim3 grid(SEQ / ROWS, JDIM);             // 8 x 96 blocks
        dim3 block(NTHR);                        // 16 waves
        lmta_fused_wmma_kernel<<<grid, block, SMEM_BYTES, stream>>>(xw, off, pred, out);
    }
}
